// UAConv_77567109366452
// MI455X (gfx1250) — matrix operand order repaired and checked by reading, not compiled, not run on hardware
//
#include <hip/hip_runtime.h>

typedef __attribute__((ext_vector_type(8)))  float        v8f;
typedef __attribute__((ext_vector_type(16))) __bf16       v16bf;
typedef __attribute__((ext_vector_type(4)))  unsigned int v4u;
typedef __attribute__((ext_vector_type(4)))  int          v4i;
typedef __attribute__((ext_vector_type(8)))  int          v8i;

#define NB 8
#define CH 64
#define HH 128
#define WW 128
#define KK 9
#define KTOT 576      /* 64*9 */
#define KCHUNKS 18    /* 576/32 */

/* ---- workspace layout (bytes) ---- */
#define ASPEC_OFF   0ull
#define ASPEC_BYTES (8ull*64*128*128*2)          /* 16,777,216  bf16 A_spec  */
#define ASPAT_OFF   (ASPEC_OFF + ASPEC_BYTES)
#define ASPAT_BYTES (8ull*9*128*128*2)           /*  2,359,296  bf16 A_spat  */
#define WPK_OFF     (ASPAT_OFF + ASPAT_BYTES)    /* 19,136,512 (32B aligned) */
#define WPK_BYTES   (4ull*18*32*16*2)            /*     73,728 packed conv_w */
#define WSPK_OFF    (WPK_OFF + WPK_BYTES)        /* 19,210,240 (32B aligned) */
#define WSPK_BYTES  (24ull*32*16*2)              /*     24,576 packed sp_w   */

/* =============== weight packing into WMMA A-operand lane layout ============
   16x32 bf16 A matrix, wave32, contiguous k order used here (B uses the same):
     lanes 0-15 : M=lane,      element e -> K = e
     lanes16-31 : M=lane-16,   element e -> K = e + 16                       */
__global__ void pack_main_w(const float* __restrict__ cw, __bf16* __restrict__ wpk) {
  const int lane = threadIdx.x;               /* 32 */
  const int blk  = blockIdx.x;                /* t*18 + kc, 72 blocks */
  const int kc   = blk % KCHUNKS;
  const int hi   = lane >> 4;
  const int m    = (blk / KCHUNKS) * 16 + (lane & 15);
#pragma unroll
  for (int e = 0; e < 16; ++e) {
    int K = kc * 32 + e + (hi ? 16 : 0);
    wpk[((size_t)blk * 32 + lane) * 16 + e] = (__bf16)cw[m * KTOT + K];
  }
}

__global__ void pack_spec_w(const float* __restrict__ w1, const float* __restrict__ w2,
                            const float* __restrict__ w3, __bf16* __restrict__ wspk) {
  const int lane = threadIdx.x;               /* 32 */
  const int blk  = blockIdx.x;                /* L*8 + t*2 + kc, 24 blocks */
  const int L  = blk >> 3;
  const float* src = (L == 0) ? w1 : ((L == 1) ? w2 : w3);
  const int kc = blk & 1;
  const int hi = lane >> 4;
  const int m  = (((blk >> 1) & 3)) * 16 + (lane & 15);
#pragma unroll
  for (int e = 0; e < 16; ++e) {
    int K = kc * 32 + e + (hi ? 16 : 0);
    wspk[((size_t)blk * 32 + lane) * 16 + e] = (__bf16)src[m * 64 + K];
  }
}

/* =============== spatial attention: 3x3 conv 64->9, two 9->9, sigmoid ===== */
__global__ __launch_bounds__(128) void ua_spat_kernel(
    const float* __restrict__ U,
    const float* __restrict__ w1, const float* __restrict__ bb1,
    const float* __restrict__ w2, const float* __restrict__ bb2,
    const float* __restrict__ w3, const float* __restrict__ bb3,
    __bf16* __restrict__ aspat)
{
  __shared__ float sw1[KK * CH * 9];          /* 20.25 KB */
  __shared__ float sw2[81], sw3[81], sb[27];
  const int tid = threadIdx.x;
  for (int i = tid; i < KK * CH * 9; i += 128) sw1[i] = w1[i];
  if (tid < 81) { sw2[tid] = w2[tid]; sw3[tid] = w3[tid]; }
  if (tid < 9)  sb[tid] = bb1[tid];
  else if (tid < 18) sb[tid] = bb2[tid - 9];
  else if (tid < 27) sb[tid] = bb3[tid - 18];
  __syncthreads();

  const int wcol = tid;                       /* 0..127 */
  const int h = blockIdx.x & 127;
  const int b = blockIdx.x >> 7;

  float g[9];
#pragma unroll
  for (int o = 0; o < 9; ++o) g[o] = sb[o];

  for (int c = 0; c < CH; ++c) {
    const float* Uc = U + ((size_t)(b * CH + c)) * (HH * WW);
#pragma unroll
    for (int kh = 0; kh < 3; ++kh) {
      int gh = h - 1 + kh;
      if ((unsigned)gh >= (unsigned)HH) continue;
#pragma unroll
      for (int kw = 0; kw < 3; ++kw) {
        int gw = wcol - 1 + kw;
        if ((unsigned)gw >= (unsigned)WW) continue;
        float uv = Uc[gh * WW + gw];
        int pos = kh * 3 + kw;
#pragma unroll
        for (int o = 0; o < 9; ++o)
          g[o] = fmaf(uv, sw1[(o * CH + c) * 9 + pos], g[o]);
      }
    }
  }
#pragma unroll
  for (int o = 0; o < 9; ++o) g[o] = fmaxf(g[o], 0.f);
  float g2[9];
#pragma unroll
  for (int o = 0; o < 9; ++o) {
    float s = sb[9 + o];
#pragma unroll
    for (int i2 = 0; i2 < 9; ++i2) s = fmaf(sw2[o * 9 + i2], g[i2], s);
    g2[o] = fmaxf(s, 0.f);
  }
#pragma unroll
  for (int o = 0; o < 9; ++o) {
    float s = sb[18 + o];
#pragma unroll
    for (int i2 = 0; i2 < 9; ++i2) s = fmaf(sw3[o * 9 + i2], g2[i2], s);
    s = 1.f / (1.f + expf(-s));
    aspat[((size_t)(b * KK + o)) * (HH * WW) + h * WW + wcol] = (__bf16)s;
  }
}

/* =============== spectral attention: 3 chained 64x64 GEMMs on WMMA ======== */
__global__ __launch_bounds__(128) void ua_spec_kernel(
    const float* __restrict__ U, const __bf16* __restrict__ wspk,
    const float* __restrict__ b1, const float* __restrict__ b2,
    const float* __restrict__ b3, __bf16* __restrict__ aspec)
{
  __shared__ float relay[4][64 * 16];         /* per-wave C/D -> B relayout */
  const int tid = threadIdx.x;
  const int wave = tid >> 5, lane = tid & 31, laneN = lane & 15;
  const int hi = lane >> 4;

  const int strip = blockIdx.x * 4 + wave;    /* 8192 strips of 16 pixels */
  const int p0  = strip * 16;
  const int bb  = p0 >> 14;
  const int hw0 = p0 & 16383;
  const float* Ub = U + (size_t)bb * (CH * 16384) + hw0;
  float* rl = relay[wave];
  const v16bf* wsp = (const v16bf*)wspk;

  const v8f vz = {0.f,0.f,0.f,0.f,0.f,0.f,0.f,0.f};
  v8f acc[4] = {vz, vz, vz, vz};

  /* layer 1: B built from global U (f32 -> bf16) */
  for (int kc = 0; kc < 2; ++kc) {
    const int klo = kc * 32 + (hi ? 16 : 0);
    v16bf bf;
#pragma unroll
    for (int e = 0; e < 16; ++e)
      bf[e] = (__bf16)Ub[(klo + e) * 16384 + laneN];
#pragma unroll
    for (int t = 0; t < 4; ++t) {
      v16bf af = wsp[((0 * 4 + t) * 2 + kc) * 32 + lane];
      acc[t] = __builtin_amdgcn_wmma_f32_16x16x32_bf16(false, af, false, bf,
                                                       (short)0, acc[t], false, false);
    }
  }
#pragma unroll
  for (int t = 0; t < 4; ++t)
#pragma unroll
    for (int i = 0; i < 8; ++i) {
      int m = t * 16 + i + (hi ? 8 : 0);
      rl[m * 16 + laneN] = fmaxf(acc[t][i] + b1[m], 0.f);
    }
  __syncthreads();

  /* layer 2 */
  v8f a2[4] = {vz, vz, vz, vz};
  for (int kc = 0; kc < 2; ++kc) {
    const int klo = kc * 32 + (hi ? 16 : 0);
    v16bf bf;
#pragma unroll
    for (int e = 0; e < 16; ++e)
      bf[e] = (__bf16)rl[(klo + e) * 16 + laneN];
#pragma unroll
    for (int t = 0; t < 4; ++t) {
      v16bf af = wsp[((1 * 4 + t) * 2 + kc) * 32 + lane];
      a2[t] = __builtin_amdgcn_wmma_f32_16x16x32_bf16(false, af, false, bf,
                                                      (short)0, a2[t], false, false);
    }
  }
  __syncthreads();
#pragma unroll
  for (int t = 0; t < 4; ++t)
#pragma unroll
    for (int i = 0; i < 8; ++i) {
      int m = t * 16 + i + (hi ? 8 : 0);
      rl[m * 16 + laneN] = fmaxf(a2[t][i] + b2[m], 0.f);
    }
  __syncthreads();

  /* layer 3 + sigmoid + bf16 store */
  v8f a3[4] = {vz, vz, vz, vz};
  for (int kc = 0; kc < 2; ++kc) {
    const int klo = kc * 32 + (hi ? 16 : 0);
    v16bf bf;
#pragma unroll
    for (int e = 0; e < 16; ++e)
      bf[e] = (__bf16)rl[(klo + e) * 16 + laneN];
#pragma unroll
    for (int t = 0; t < 4; ++t) {
      v16bf af = wsp[((2 * 4 + t) * 2 + kc) * 32 + lane];
      a3[t] = __builtin_amdgcn_wmma_f32_16x16x32_bf16(false, af, false, bf,
                                                      (short)0, a3[t], false, false);
    }
  }
#pragma unroll
  for (int t = 0; t < 4; ++t)
#pragma unroll
    for (int i = 0; i < 8; ++i) {
      int m = t * 16 + i + (hi ? 8 : 0);
      float v = a3[t][i] + b3[m];
      v = 1.f / (1.f + expf(-v));
      aspec[((size_t)bb * 64 + m) * 16384 + hw0 + laneN] = (__bf16)v;
    }
}

/* =============== TDM 2D tile -> LDS (bf16 strided rows) =================== */
#if __has_builtin(__builtin_amdgcn_tensor_load_to_lds)
__device__ __forceinline__ void tdm_load_2d(unsigned lds_addr, const void* gaddr,
                                            unsigned tile0, unsigned tile1,
                                            unsigned long long stride0_elems)
{
  unsigned long long ga = (unsigned long long)(uintptr_t)gaddr;
  v4u g0;
  g0[0] = 1u;                                              /* count=1, user D# */
  g0[1] = lds_addr;                                        /* lds_addr[63:32]  */
  g0[2] = (unsigned)(ga & 0xFFFFFFFFu);                    /* global_addr lo   */
  g0[3] = (unsigned)((ga >> 32) & 0x1FFFFFFu) | (2u << 30);/* addr hi + type=2 */

  const unsigned long long dim0 = 1ull << 20, dim1 = 1ull << 20; /* no OOB clip */
  unsigned long long q0 = (1ull << 16)                      /* data_size=1 (2B) */
                        | ((dim0 & 0xFFFFull) << 48);
  unsigned long long q1 = ((dim0 >> 16) & 0xFFFFull)
                        | ((dim1 & 0xFFFFFFFFull) << 16)
                        | ((unsigned long long)(tile0 & 0xFFFFu) << 48);
  unsigned long long q2 = (unsigned long long)(tile1 & 0xFFFFu)
                        | ((stride0_elems & 0xFFFFFFFFull) << 32);
  unsigned long long q3 = ((stride0_elems >> 32) & 0xFFFFull);
  v8i g1;
  g1[0] = (int)(unsigned)q0; g1[1] = (int)(unsigned)(q0 >> 32);
  g1[2] = (int)(unsigned)q1; g1[3] = (int)(unsigned)(q1 >> 32);
  g1[4] = (int)(unsigned)q2; g1[5] = (int)(unsigned)(q2 >> 32);
  g1[6] = (int)(unsigned)q3; g1[7] = (int)(unsigned)(q3 >> 32);
  v4i z4 = {0, 0, 0, 0};
#if defined(__clang_major__) && __clang_major__ >= 23
  v8i z8 = {0, 0, 0, 0, 0, 0, 0, 0};
  __builtin_amdgcn_tensor_load_to_lds(g0, g1, z4, z4, z8, 0);
#else
  __builtin_amdgcn_tensor_load_to_lds(g0, g1, z4, z4, 0);
#endif
}
#endif

/* =============== fused main: im2col * A_spec * A_spat -> WMMA GEMM ======== */
__global__ __launch_bounds__(128) void ua_main_kernel(
    const float* __restrict__ x, const __bf16* __restrict__ aspec,
    const __bf16* __restrict__ aspat, const __bf16* __restrict__ wpk,
    float* __restrict__ out)
{
  __shared__ __bf16 xtile[CH * 3 * 68];       /* 64c x 3 rows x 66(+2) cols  */
  __shared__ __bf16 s_aspec[CH * 64];
  __shared__ __bf16 s_aspat[KK * 64];

  const int tid = threadIdx.x;
  const int bid = blockIdx.x;
  const int seg = bid & 1;
  const int h   = (bid >> 1) & 127;
  const int b   = bid >> 8;
  const int w0  = seg * 64;

  __builtin_prefetch(wpk, 0, 1);              /* global_prefetch_b8 on weights */

  /* stage padded x tile, f32 -> bf16 */
  for (int i = tid; i < CH * 3 * 68; i += 128) {
    int col = i % 68;
    int rc  = i / 68;
    int r   = rc % 3;
    int c   = rc / 3;
    int gh = h - 1 + r;
    int gw = w0 - 1 + col;
    float v = 0.f;
    if (col < 66 && (unsigned)gh < (unsigned)HH && (unsigned)gw < (unsigned)WW)
      v = x[((b * CH + c) * HH + gh) * WW + gw];
    xtile[i] = (__bf16)v;
  }

  /* stage attention rows: TDM async tensor loads (wave 0 issues) */
#if __has_builtin(__builtin_amdgcn_tensor_load_to_lds)
  if (tid < 32) {
    tdm_load_2d((unsigned)(uintptr_t)&s_aspec[0],
                aspec + ((size_t)b * CH) * 16384 + h * 128 + w0,
                64, 64, 16384ull);
    tdm_load_2d((unsigned)(uintptr_t)&s_aspat[0],
                aspat + ((size_t)b * KK) * 16384 + h * 128 + w0,
                64, KK, 16384ull);
    __builtin_amdgcn_s_wait_tensorcnt(0);
  }
#else
  for (int i = tid; i < CH * 64; i += 128) {
    int c = i >> 6, cc = i & 63;
    s_aspec[i] = aspec[((size_t)b * CH + c) * 16384 + h * 128 + w0 + cc];
  }
  for (int i = tid; i < KK * 64; i += 128) {
    int p = i >> 6, cc = i & 63;
    s_aspat[i] = aspat[((size_t)b * KK + p) * 16384 + h * 128 + w0 + cc];
  }
#endif
  __syncthreads();

  const int wave = tid >> 5, lane = tid & 31, laneN = lane & 15;
  const int hi = lane >> 4;
  const int lc = wave * 16 + laneN;           /* local pixel column 0..63 */

  const v8f vz = {0.f,0.f,0.f,0.f,0.f,0.f,0.f,0.f};
  v8f acc[4] = {vz, vz, vz, vz};
  const v16bf* wm = (const v16bf*)wpk;

  for (int kc = 0; kc < KCHUNKS; ++kc) {
    const int klo = kc * 32 + (hi ? 16 : 0);
    v16bf bf;
#pragma unroll
    for (int e = 0; e < 16; ++e) {
      int K = klo + e;
      int c = (K * 7282) >> 16;               /* K/9 for K<576 */
      int p = K - 9 * c;
      int dh = p / 3, dw = p % 3;
      float xv = (float)xtile[(c * 3 + dh) * 68 + lc + dw];
      float as = (float)s_aspec[c * 64 + lc];
      float ap = (float)s_aspat[p * 64 + lc];
      bf[e] = (__bf16)(xv * as * ap);
    }
#pragma unroll
    for (int t = 0; t < 4; ++t) {
      v16bf af = wm[(t * KCHUNKS + kc) * 32 + lane];
      acc[t] = __builtin_amdgcn_wmma_f32_16x16x32_bf16(false, af, false, bf,
                                                       (short)0, acc[t], false, false);
    }
  }

#pragma unroll
  for (int t = 0; t < 4; ++t)
#pragma unroll
    for (int i = 0; i < 8; ++i) {
      int m = t * 16 + i + (hi ? 8 : 0);
      out[((b * 64 + m) * HH + h) * WW + w0 + lc] = acc[t][i];
    }
}

/* ========================= host launcher ================================== */
extern "C" void kernel_launch(void* const* d_in, const int* in_sizes, int n_in,
                              void* d_out, int out_size, void* d_ws, size_t ws_size,
                              hipStream_t stream) {
  const float* x      = (const float*)d_in[0];
  const float* U      = (const float*)d_in[1];
  const float* sg1_w  = (const float*)d_in[2];
  const float* sg1_b  = (const float*)d_in[3];
  const float* sg2_w  = (const float*)d_in[4];
  const float* sg2_b  = (const float*)d_in[5];
  const float* sg3_w  = (const float*)d_in[6];
  const float* sg3_b  = (const float*)d_in[7];
  const float* sp1_w  = (const float*)d_in[8];
  const float* sp1_b  = (const float*)d_in[9];
  const float* sp2_w  = (const float*)d_in[10];
  const float* sp2_b  = (const float*)d_in[11];
  const float* sp3_w  = (const float*)d_in[12];
  const float* sp3_b  = (const float*)d_in[13];
  const float* conv_w = (const float*)d_in[14];

  char* ws = (char*)d_ws;
  __bf16* aspec = (__bf16*)(ws + ASPEC_OFF);
  __bf16* aspat = (__bf16*)(ws + ASPAT_OFF);
  __bf16* wpk   = (__bf16*)(ws + WPK_OFF);
  __bf16* wspk  = (__bf16*)(ws + WSPK_OFF);
  float* out = (float*)d_out;

  pack_main_w<<<4 * KCHUNKS, 32, 0, stream>>>(conv_w, wpk);
  pack_spec_w<<<24, 32, 0, stream>>>(sp1_w, sp2_w, sp3_w, wspk);
  ua_spat_kernel<<<NB * HH, 128, 0, stream>>>(U, sg1_w, sg1_b, sg2_w, sg2_b,
                                              sg3_w, sg3_b, aspat);
  ua_spec_kernel<<<2048, 128, 0, stream>>>(U, wspk, sp1_b, sp2_b, sp3_b, aspec);
  ua_main_kernel<<<2048, 128, 0, stream>>>(x, aspec, aspat, wpk, out);
}
